// TriangleMultiplicativeUpdate_86199993630898
// MI455X (gfx1250) — matrix operand order repaired and checked by reading, not compiled, not run on hardware
//
#include <hip/hip_runtime.h>

#define NTOK 512
#define DCH  128
#define NN   (NTOK * NTOK)

typedef __attribute__((ext_vector_type(16))) __bf16 v16bf;
typedef __attribute__((ext_vector_type(8)))  __bf16 v8bf;
typedef __attribute__((ext_vector_type(8)))  float  v8f;
typedef __attribute__((ext_vector_type(4)))  float  v4f;
typedef unsigned int v4u __attribute__((ext_vector_type(4)));
typedef int          v8i __attribute__((ext_vector_type(8)));
typedef int          v4i __attribute__((ext_vector_type(4)));

// ---------- fragment loaders ----------
// A-fragment (16x32 bf16): lane holds row M; elements 0..15 = K base..base+15
// (base = 0 for lanes 0-15, 16 for 16-31).
__device__ __forceinline__ v16bf ld_afrag(const __bf16* p) {
    v8bf lo = *(const v8bf*)p;
    v8bf hi = *(const v8bf*)(p + 8);
    v16bf r;
#pragma unroll
    for (int i = 0; i < 8; ++i) { r[i] = lo[i]; r[i + 8] = hi[i]; }
    return r;
}
// B-fragment (32x16 bf16): lane holds column N; 16 contiguous K values.
__device__ __forceinline__ v16bf ld_bfrag(const __bf16* p) {
    v8bf lo = *(const v8bf*)p;
    v8bf hi = *(const v8bf*)(p + 8);
    v16bf r;
#pragma unroll
    for (int i = 0; i < 8; ++i) { r[i] = lo[i]; r[i + 8] = hi[i]; }
    return r;
}

__device__ __forceinline__ float sigmoidf(float x) {
    return 1.0f / (1.0f + __expf(-x));
}

// ---------- TDM: issue a 64x64 bf16 tile load (2D tensor 512x512) ----------
// D# built per CDNA5 ISA 8.3/8.4: count=1, type=2(image), data_size=2B,
// pad_enable with pad_interval=32 DWORDs (one 128B tile row) and
// pad_amount=4 DWORDs (16B) -> LDS row stride 72 bf16 (bank-friendly).
// This toolchain exposes the 6-arg builtin:
//   (uint32x4 g0, int32x8 g1, int32x4, int32x4, int32x8, i32 cpol)
#if __has_builtin(__builtin_amdgcn_tensor_load_to_lds)
#define TMU_HAVE_TDM 1
__device__ __forceinline__ void tdm_load_tile64(const __bf16* gtile,
                                                unsigned int lds_off) {
    unsigned long long ga = (unsigned long long)(size_t)gtile;
    v4u g0;
    g0[0] = 1u;                                   // count=1 (valid descriptor)
    g0[1] = lds_off;                              // lds_addr (bytes)
    g0[2] = (unsigned int)ga;                     // global_addr[31:0]
    g0[3] = (unsigned int)((ga >> 32) & 0x1FFFFFFu) | (2u << 30); // [56:32] | type=2
    v8i g1;
    g1[0] = (1 << 16)     // data_size = 2 bytes
          | (1 << 20)     // pad_enable
          | (4 << 22)     // pad_interval: 2^(4+1)=32 DWORDs
          | (3 << 25);    // pad_amount: 3+1 = 4 DWORDs
    g1[1] = (NTOK & 0xFFFF) << 16;   // tensor_dim0[15:0] (atomic_barrier_addr=0)
    g1[2] = (NTOK & 0xFFFF) << 16;   // tensor_dim0[31:16]=0 | tensor_dim1[15:0]
    g1[3] = (64 << 16);              // tensor_dim1[31:16]=0 | tile_dim0=64
    g1[4] = 64;                      // tile_dim1=64, tile_dim2=0
    g1[5] = NTOK;                    // tensor_dim0_stride[31:0] = 512
    g1[6] = 0;                       // stride0 hi | tensor_dim1_stride lo (unused, 2D)
    g1[7] = 0;
    v4i z4 = {0, 0, 0, 0};           // groups 2/3 unused (2D tensor)
    v8i z8 = {0, 0, 0, 0, 0, 0, 0, 0};
    __builtin_amdgcn_tensor_load_to_lds(g0, g1, z4, z4, z8, 0);
}
#else
#define TMU_HAVE_TDM 0
#endif

__device__ __forceinline__ void tmu_wait_tensor0() {
#if __has_builtin(__builtin_amdgcn_s_wait_tensorcnt)
    __builtin_amdgcn_s_wait_tensorcnt(0);
#else
    asm volatile("s_wait_tensorcnt 0x0" ::: "memory");
#endif
}

// ---------- stage 0: weight conversion fp32 -> bf16 ----------
__global__ void tmu_convert_bf16(const float* __restrict__ src,
                                 unsigned short* __restrict__ dst, int n) {
    int i = blockIdx.x * blockDim.x + threadIdx.x;
    if (i < n) {
        __bf16 v = (__bf16)src[i];
        dst[i] = *(unsigned short*)&v;
    }
}

// ---------- stage 1: fused LayerNorm(in) + projections (wp, wg, w_gate) ----------
// Block: 256 threads, 128 rows of (i,j).  Outputs (all channel-major bf16):
//   a_t[d][row], b_t[d][row], gate_t[d][row]
__global__ __launch_bounds__(256, 1)
void tmu_ln_proj(const float* __restrict__ z, const float* __restrict__ mask,
                 const float* __restrict__ lnw, const float* __restrict__ lnb,
                 const unsigned short* __restrict__ wpb_,
                 const unsigned short* __restrict__ wgb_,
                 const unsigned short* __restrict__ wgateb_,
                 unsigned short* __restrict__ a_t_,
                 unsigned short* __restrict__ b_t_,
                 unsigned short* __restrict__ gate_) {
    const __bf16* wpb    = (const __bf16*)wpb_;
    const __bf16* wgb    = (const __bf16*)wgb_;
    const __bf16* wgateb = (const __bf16*)wgateb_;
    __bf16* a_t    = (__bf16*)a_t_;
    __bf16* b_t    = (__bf16*)b_t_;
    __bf16* gate_t = (__bf16*)gate_;

    __shared__ __bf16 hs[128 * 136];  // 128 rows x 128 ch, stride 136
    __shared__ float  msk[128];

    const int t  = threadIdx.x;
    const int rb = blockIdx.x * 128;

    // -------- phase 1: LayerNorm of 128 rows into LDS (bf16) --------
    {
        const int rl = t >> 1, half = t & 1;
        const float* zp = z + (size_t)(rb + rl) * DCH + half * 64;
        float s = 0.f, ss = 0.f;
#pragma unroll
        for (int c = 0; c < 64; c += 4) {
            v4f v = *(const v4f*)(zp + c);
            s  += v.x + v.y + v.z + v.w;
            ss += v.x * v.x + v.y * v.y + v.z * v.z + v.w * v.w;
        }
        s  += __shfl_xor(s, 1, 32);
        ss += __shfl_xor(ss, 1, 32);
        const float mu   = s * (1.0f / DCH);
        const float var  = ss * (1.0f / DCH) - mu * mu;
        const float rstd = rsqrtf(var + 1e-5f);
#pragma unroll 8
        for (int c = 0; c < 64; ++c) {
            const int dd = half * 64 + c;
            const float hv = (zp[c] - mu) * rstd * lnw[dd] + lnb[dd];
            hs[rl * 136 + dd] = (__bf16)hv;
        }
        if (t < 128) msk[t] = mask[rb + t];
    }
    __syncthreads();

    // -------- phase 2: WMMA projections --------
    const int lane  = t & 31;
    const int w     = t >> 5;       // wave id 0..7 -> 16-row tile
    const int mr    = w * 16;
    const int jn    = lane & 15;
    const int kh    = lane >> 4;
    const int mbase = kh * 8;

    v16bf af[4];
#pragma unroll
    for (int c = 0; c < 4; ++c)
        af[c] = ld_afrag(&hs[(mr + jn) * 136 + c * 32 + kh * 16]);

    // p/g pairs: 16 column tiles covering e = 0..255
    for (int tt = 0; tt < 16; ++tt) {
        // burst-load all 8 B-fragments, then run the WMMA chain (single wait)
        v16bf bp[4], bg[4];
#pragma unroll
        for (int c = 0; c < 4; ++c) {
            bp[c] = ld_bfrag(wpb + (size_t)(tt * 16 + jn) * DCH + c * 32 + kh * 16);
            bg[c] = ld_bfrag(wgb + (size_t)(tt * 16 + jn) * DCH + c * 32 + kh * 16);
        }
        v8f accp = {};
        v8f accg = {};
#pragma unroll
        for (int c = 0; c < 4; ++c)
            accp = __builtin_amdgcn_wmma_f32_16x16x32_bf16(false, af[c], false, bp[c],
                                                           (short)0, accp, false, false);
#pragma unroll
        for (int c = 0; c < 4; ++c)
            accg = __builtin_amdgcn_wmma_f32_16x16x32_bf16(false, af[c], false, bg[c],
                                                           (short)0, accg, false, false);
        const int e = tt * 16 + jn;
        v8bf outv;
#pragma unroll
        for (int r = 0; r < 8; ++r) {
            const int rl = mr + mbase + r;
            const float ab = accp[r] * sigmoidf(accg[r]) * msk[rl];
            outv[r] = (__bf16)ab;
        }
        const size_t rowg = (size_t)rb + mr + mbase;
        __bf16* dst = (e < DCH) ? (a_t + (size_t)e * NN + rowg)
                                : (b_t + (size_t)(e - DCH) * NN + rowg);
        *(v8bf*)dst = outv;  // 8 consecutive rows, 16B store
    }

    // gate: 8 column tiles covering e = 0..127, stored channel-major
    for (int tt = 0; tt < 8; ++tt) {
        v16bf bw[4];
#pragma unroll
        for (int c = 0; c < 4; ++c)
            bw[c] = ld_bfrag(wgateb + (size_t)(tt * 16 + jn) * DCH + c * 32 + kh * 16);
        v8f acc = {};
#pragma unroll
        for (int c = 0; c < 4; ++c)
            acc = __builtin_amdgcn_wmma_f32_16x16x32_bf16(false, af[c], false, bw[c],
                                                          (short)0, acc, false, false);
        const int e = tt * 16 + jn;
        v8bf gv;
#pragma unroll
        for (int r = 0; r < 8; ++r) gv[r] = (__bf16)sigmoidf(acc[r]);
        const size_t rowg = (size_t)rb + mr + mbase;
        *(v8bf*)(gate_t + (size_t)e * NN + rowg) = gv;
    }
}

// ---------- stage 2: triangle contraction, batched over d ----------
// x_t[d][i][j] = sum_k a_t[d][i][k] * b_t[d][j][k]
// 64x64 output tile per block, K staged 64-wide through LDS, double-buffered
// via Tensor Data Mover (wave 0 issues descriptors, TENSORcnt + barrier sync).
__global__ __launch_bounds__(256, 1)
void tmu_triangle(const unsigned short* __restrict__ a_t_,
                  const unsigned short* __restrict__ b_t_,
                  float* __restrict__ x_t) {
    const __bf16* a_t = (const __bf16*)a_t_;
    const __bf16* b_t = (const __bf16*)b_t_;

    __shared__ __bf16 As[2][64 * 72];  // 64 rows x 64 K, stride 72 (TDM-padded)
    __shared__ __bf16 Bs[2][64 * 72];

    const int d  = blockIdx.y;
    const int i0 = (blockIdx.x & 7) * 64;
    const int j0 = (blockIdx.x >> 3) * 64;

    const int t    = threadIdx.x;
    const int lane = t & 31;
    const int w    = t >> 5;
    const int mt   = w & 3;            // row tile 0..3
    const int ntg  = (w >> 2) * 2;     // two col tiles per wave
    const int jn   = lane & 15;
    const int kh   = lane >> 4;

    const __bf16* Ag = a_t + (size_t)d * NN + (size_t)i0 * NTOK;
    const __bf16* Bg = b_t + (size_t)d * NN + (size_t)j0 * NTOK;

#if TMU_HAVE_TDM
    typedef __attribute__((address_space(3))) void* lds_vp;
    const unsigned int lAs0 = (unsigned int)(size_t)(lds_vp)&As[0][0];
    const unsigned int lAs1 = (unsigned int)(size_t)(lds_vp)&As[1][0];
    const unsigned int lBs0 = (unsigned int)(size_t)(lds_vp)&Bs[0][0];
    const unsigned int lBs1 = (unsigned int)(size_t)(lds_vp)&Bs[1][0];
    if (w == 0) {                       // prologue: stage 0 into buffer 0
        tdm_load_tile64(Ag, lAs0);
        tdm_load_tile64(Bg, lBs0);
    }
#else
    const int lr = t >> 2, lq = t & 3; // cooperative fallback loader indices
    {
        *(v8bf*)&As[0][lr * 72 + lq * 16]     = *(const v8bf*)(Ag + (size_t)lr * NTOK + lq * 16);
        *(v8bf*)&As[0][lr * 72 + lq * 16 + 8] = *(const v8bf*)(Ag + (size_t)lr * NTOK + lq * 16 + 8);
        *(v8bf*)&Bs[0][lr * 72 + lq * 16]     = *(const v8bf*)(Bg + (size_t)lr * NTOK + lq * 16);
        *(v8bf*)&Bs[0][lr * 72 + lq * 16 + 8] = *(const v8bf*)(Bg + (size_t)lr * NTOK + lq * 16 + 8);
    }
#endif

    v8f acc0 = {}, acc1 = {};
    for (int kc = 0; kc < NTOK; kc += 64) {
        const int buf = (kc >> 6) & 1;
#if TMU_HAVE_TDM
        if (w == 0) tmu_wait_tensor0();   // our TDM loads into `buf` complete
        __syncthreads();                  // all waves: buf ready, buf^1 drained
        if (w == 0 && kc + 64 < NTOK) {   // prefetch next stage into other buffer
            tdm_load_tile64(Ag + (kc + 64), buf ? lAs0 : lAs1);
            tdm_load_tile64(Bg + (kc + 64), buf ? lBs0 : lBs1);
        }
#else
        __syncthreads();
        if (kc + 64 < NTOK) {
            const int nb = buf ^ 1;
            const __bf16* An = Ag + (kc + 64);
            const __bf16* Bn = Bg + (kc + 64);
            *(v8bf*)&As[nb][lr * 72 + lq * 16]     = *(const v8bf*)(An + (size_t)lr * NTOK + lq * 16);
            *(v8bf*)&As[nb][lr * 72 + lq * 16 + 8] = *(const v8bf*)(An + (size_t)lr * NTOK + lq * 16 + 8);
            *(v8bf*)&Bs[nb][lr * 72 + lq * 16]     = *(const v8bf*)(Bn + (size_t)lr * NTOK + lq * 16);
            *(v8bf*)&Bs[nb][lr * 72 + lq * 16 + 8] = *(const v8bf*)(Bn + (size_t)lr * NTOK + lq * 16 + 8);
        }
#endif
#pragma unroll
        for (int kk = 0; kk < 2; ++kk) {  // two 32-wide K chunks per stage
            v16bf afr = ld_afrag(&As[buf][(mt * 16 + jn) * 72 + kk * 32 + kh * 16]);
            v16bf b0  = ld_bfrag(&Bs[buf][((ntg + 0) * 16 + jn) * 72 + kk * 32 + kh * 16]);
            v16bf b1  = ld_bfrag(&Bs[buf][((ntg + 1) * 16 + jn) * 72 + kk * 32 + kh * 16]);
            acc0 = __builtin_amdgcn_wmma_f32_16x16x32_bf16(false, afr, false, b0,
                                                           (short)0, acc0, false, false);
            acc1 = __builtin_amdgcn_wmma_f32_16x16x32_bf16(false, afr, false, b1,
                                                           (short)0, acc1, false, false);
        }
#if !TMU_HAVE_TDM
        __syncthreads();
#endif
    }

    const int mbase = kh * 8;
#pragma unroll
    for (int r = 0; r < 8; ++r) {
        const size_t ig = (size_t)(i0 + mt * 16 + mbase + r);
        x_t[(size_t)d * NN + ig * NTOK + j0 + (ntg + 0) * 16 + jn] = acc0[r];
        x_t[(size_t)d * NN + ig * NTOK + j0 + (ntg + 1) * 16 + jn] = acc1[r];
    }
}

// ---------- stage 3: LayerNorm(out) + w_out projection + gate ----------
__global__ __launch_bounds__(256, 1)
void tmu_out(const float* __restrict__ x_t,
             const unsigned short* __restrict__ gate_,
             const float* __restrict__ lnw, const float* __restrict__ lnb,
             const unsigned short* __restrict__ woutb_,
             float* __restrict__ out) {
    const __bf16* gate_t = (const __bf16*)gate_;
    const __bf16* woutb  = (const __bf16*)woutb_;

    __shared__ float  xs[128 * 68];   // [channel][row], stride 68 (16B aligned)
    __shared__ __bf16 hs[64 * 136];   // normalized rows, bf16

    const int t  = threadIdx.x;
    const int r0 = blockIdx.x * 64;

    // phase 1: stage x (channel-major) into LDS
    {
        const int d = t >> 1, half = t & 1;
        const float* src = x_t + (size_t)d * NN + r0 + half * 32;
        float* dst = &xs[d * 68 + half * 32];
#pragma unroll
        for (int c = 0; c < 32; c += 4)
            *(v4f*)(dst + c) = *(const v4f*)(src + c);
    }
    __syncthreads();

    // phase 2: LayerNorm per row over 128 channels
    if (t < 64) {
        float s = 0.f, ss = 0.f;
#pragma unroll 8
        for (int d = 0; d < DCH; ++d) {
            const float v = xs[d * 68 + t];
            s += v;
            ss += v * v;
        }
        const float mu   = s * (1.0f / DCH);
        const float var  = ss * (1.0f / DCH) - mu * mu;
        const float rstd = rsqrtf(var + 1e-5f);
#pragma unroll 4
        for (int d = 0; d < DCH; ++d) {
            const float hv = (xs[d * 68 + t] - mu) * rstd * lnw[d] + lnb[d];
            hs[t * 136 + d] = (__bf16)hv;
        }
    }
    __syncthreads();

    // phase 3: WMMA with w_out, apply gate, write output
    const int lane  = t & 31;
    const int w     = t >> 5;
    const int mt    = w & 3;
    const int ng    = (w >> 2) * 4;
    const int jn    = lane & 15;
    const int kh    = lane >> 4;
    const int mbase = kh * 8;

    v16bf af[4];
#pragma unroll
    for (int c = 0; c < 4; ++c)
        af[c] = ld_afrag(&hs[(mt * 16 + jn) * 136 + c * 32 + kh * 16]);

    for (int q = 0; q < 4; ++q) {
        const int nt = ng + q;
        v16bf bf[4];
#pragma unroll
        for (int c = 0; c < 4; ++c)
            bf[c] = ld_bfrag(woutb + (size_t)(nt * 16 + jn) * DCH + c * 32 + kh * 16);
        v8f acc = {};
#pragma unroll
        for (int c = 0; c < 4; ++c)
            acc = __builtin_amdgcn_wmma_f32_16x16x32_bf16(false, af[c], false, bf[c],
                                                          (short)0, acc, false, false);
        const int e = nt * 16 + jn;
        const size_t rowg0 = (size_t)r0 + mt * 16 + mbase;
        v8bf gv = *(const v8bf*)(gate_t + (size_t)e * NN + rowg0);  // 8 rows, 16B
#pragma unroll
        for (int r = 0; r < 8; ++r)
            out[(rowg0 + r) * DCH + e] = acc[r] * (float)gv[r];
    }
}

// ---------- host launcher ----------
extern "C" void kernel_launch(void* const* d_in, const int* in_sizes, int n_in,
                              void* d_out, int out_size, void* d_ws, size_t ws_size,
                              hipStream_t stream) {
    const float* z      = (const float*)d_in[0];
    const float* mask   = (const float*)d_in[1];
    const float* nin_w  = (const float*)d_in[2];
    const float* nin_b  = (const float*)d_in[3];
    const float* wp     = (const float*)d_in[4];
    const float* wg     = (const float*)d_in[5];
    const float* nout_w = (const float*)d_in[6];
    const float* nout_b = (const float*)d_in[7];
    const float* w_out  = (const float*)d_in[8];
    const float* w_gate = (const float*)d_in[9];
    float* out = (float*)d_out;

    char* ws = (char*)d_ws;
    unsigned short* wpb    = (unsigned short*)(ws + 0);            // 256*128 bf16
    unsigned short* wgb    = (unsigned short*)(ws + 65536);        // 256*128 bf16
    unsigned short* wgateb = (unsigned short*)(ws + 131072);       // 128*128 bf16
    unsigned short* woutb  = (unsigned short*)(ws + 163840);       // 128*128 bf16
    unsigned short* a_t    = (unsigned short*)(ws + 196608);       // 128*NN bf16
    unsigned short* b_t    = (unsigned short*)(ws + 196608 + (size_t)DCH * NN * 2);
    unsigned short* gate   = (unsigned short*)(ws + 196608 + (size_t)DCH * NN * 4);
    float*          x_t    = (float*)         (ws + 196608 + (size_t)DCH * NN * 6);

    tmu_convert_bf16<<<(256 * 128 + 255) / 256, 256, 0, stream>>>(wp, wpb, 256 * 128);
    tmu_convert_bf16<<<(256 * 128 + 255) / 256, 256, 0, stream>>>(wg, wgb, 256 * 128);
    tmu_convert_bf16<<<(128 * 128 + 255) / 256, 256, 0, stream>>>(w_gate, wgateb, 128 * 128);
    tmu_convert_bf16<<<(128 * 128 + 255) / 256, 256, 0, stream>>>(w_out, woutb, 128 * 128);

    tmu_ln_proj<<<NN / 128, 256, 0, stream>>>(z, mask, nin_w, nin_b,
                                              wpb, wgb, wgateb, a_t, b_t, gate);

    dim3 gridC(64, DCH);
    tmu_triangle<<<gridC, 256, 0, stream>>>(a_t, b_t, x_t);

    tmu_out<<<NN / 64, 256, 0, stream>>>(x_t, gate, nout_w, nout_b, woutb, out);
}
